// WarpNet_35098472743075
// MI455X (gfx1250) — matrix operand order repaired and checked by reading, not compiled, not run on hardware
//
#include <hip/hip_runtime.h>
#include <hip/hip_bf16.h>
#include <math.h>

// ---------------------------------------------------------------------------
// WarpNet for MI455X (gfx1250, wave32).
// Heavy work = two 3x3 convs (58 GFLOP) -> implicit GEMM via
// v_wmma_f32_16x16x32_bf16 (bf16 in, f32 accumulate).
// Workgroup = one image row; input rows staged in LDS per 64-channel chunk and
// shared by all M-tile waves (8x cut in L2 fragment traffic); each wave is
// register-blocked 2 M-tiles x 4 N-tiles.
// Everything else (<1 GFLOP, precision-sensitive softmaxes) = fp32 VALU.
// ---------------------------------------------------------------------------

typedef __attribute__((ext_vector_type(16))) __bf16 v16bf;
typedef __attribute__((ext_vector_type(8)))  __bf16 v8bf;
typedef __attribute__((ext_vector_type(8)))  float  v8f;

#define HH   64
#define WWD  64
#define HWP  4096          // 64*64
#define CIN  512
#define NIMG 4
#define KTOT (9 * CIN)     // 4608
#define C2   128
#define CS   256
#define NCLS 124
#define XOUT (2 * NCLS * HWP)  // 1,015,808 floats (log-softmax output)

#define SHUF16(lo, hi) __builtin_shufflevector(lo, hi, 0, 1, 2, 3, 4, 5, 6, 7, \
                                                       8, 9, 10, 11, 12, 13, 14, 15)

// ---------------- prep: input NCHW f32 -> NHWC bf16 ------------------------
__global__ void k_prep_x(const float* __restrict__ x, __hip_bfloat16* __restrict__ xbf) {
    int tid = blockIdx.x * blockDim.x + threadIdx.x;   // NIMG*HWP*CIN = 8,388,608
    int c = tid & (CIN - 1);
    int t = tid >> 9;
    int w = t & 63; t >>= 6;
    int h = t & 63;
    int n = t >> 6;
    xbf[tid] = __float2bfloat16(x[((n * CIN + c) * HH + h) * WWD + w]);
}

// ---------------- prep: weights OIHW f32 -> [O][j][c] bf16 -----------------
__global__ void k_prep_w(const float* __restrict__ W, __hip_bfloat16* __restrict__ wbf, int Cout) {
    int tid = blockIdx.x * blockDim.x + threadIdx.x;   // Cout*KTOT
    if (tid >= Cout * KTOT) return;
    int c = tid & (CIN - 1);
    int t = tid >> 9;
    int j = t % 9;
    int o = t / 9;
    wbf[tid] = __float2bfloat16(W[(o * CIN + c) * 9 + j]);
}

// ---------------- prep: fold BN into scale/bias ----------------------------
__global__ void k_prep_bn(const float* __restrict__ g, const float* __restrict__ b,
                          const float* __restrict__ m, const float* __restrict__ v,
                          float* __restrict__ scale, float* __restrict__ bias, int C) {
    int i = blockIdx.x * blockDim.x + threadIdx.x;
    if (i >= C) return;
    float s = g[i] * rsqrtf(v[i] + 1e-5f);
    scale[i] = s;
    bias[i]  = b[i] - m[i] * s;
}

// ---------------- implicit-GEMM conv3x3 + BN + ReLU (WMMA bf16) ------------
// Workgroup <-> one image row (n, y). COUT/32 waves; wave w owns out-channels
// [w*32, w*32+32) as 2 M-tiles, x 4 N-tiles covering the 64-pixel row.
// K = 9*512 processed as 8 chunks of 64 channels; per chunk the 3 input rows
// are staged into LDS (zero-filled at borders) and shared by all waves.
// A fragment (16-bit 16x32): lane M = lane&15; K halves = two contiguous
//   8-half runs at [kb, kb+8) and [kb+16, kb+24), kb = (lane>=16)?8:0.
// B fragment (32x16): lane N = lane&15; K = same two 8-half runs as A, at
//   [kb, kb+8) and [kb+16, kb+24); read from
//   LDS as two 16B runs (padded pixel stride 72 halves = 36 dwords -> 16
//   distinct banks across the 16-lane pixel sweep).
template <int COUT>
__global__ void k_conv_wmma(const __bf16* __restrict__ xbf, const __bf16* __restrict__ wbf,
                            const float* __restrict__ scale, const float* __restrict__ bias,
                            float* __restrict__ out) {
    constexpr int WAVES = COUT / 32;
    constexpr int BT    = WAVES * 32;
    constexpr int PSTR  = 72;                    // halves per staged pixel (144 B)
    __shared__ __bf16 sb[3 * 64 * PSTR];         // 27,648 B

    int rt = blockIdx.x;                         // 0..255 : image-row (n*64 + y)
    int n  = rt >> 6;
    int y  = rt & 63;
    int lane = threadIdx.x & 31;
    int o0   = (threadIdx.x >> 5) * 32;
    int lm = lane & 15;
    int hi = (lane >> 4) & 1;

    const __bf16* arow0 = wbf + (size_t)(o0 + lm) * KTOT + hi * 8;
    const __bf16* arow1 = arow0 + (size_t)16 * KTOT;
    const __bf16* xrow  = xbf + (size_t)(n * HWP) * CIN;

    v8f acc[8] = {};                             // [m*4 + t]
    for (int cc = 0; cc < 8; ++cc) {
        int cbase = cc * 64;
        __syncthreads();
        // ---- stage 3 rows x 64 px x 64 ch into LDS (16B packets) ----
        for (int i = threadIdx.x; i < 3 * 64 * 8; i += BT) {
            int dyi = i >> 9;                    // 0..2
            int rem = i & 511;
            int p   = rem >> 3;                  // pixel 0..63
            int pk  = rem & 7;                   // 8-half packet 0..7
            int yy  = y + dyi - 1;
            v8bf v = {};
            if ((unsigned)yy < 64u)
                v = *(const v8bf*)(xrow + (size_t)(yy * 64 + p) * CIN + cbase + pk * 8);
            *(v8bf*)(&sb[(dyi * 64 + p) * PSTR + pk * 8]) = v;
        }
        __syncthreads();

        // prefetch next chunk (one 128B line = one pixel's next 64 channels)
        if (cc < 7) {
            for (int i = threadIdx.x; i < 3 * 64; i += BT) {
                int dyi = i >> 6;
                int p   = i & 63;
                int yy  = y + dyi - 1;
                if ((unsigned)yy < 64u)
                    __builtin_prefetch((const void*)(xrow + (size_t)(yy * 64 + p) * CIN +
                                                     cbase + 64), 0, 0);
            }
        }

        // ---- consume: 9 taps x 2 K-steps of 32 ----
        for (int j = 0; j < 9; ++j) {
            int dyi = j / 3;
            int dx  = j - dyi * 3 - 1;
            if ((unsigned)(y + dyi - 1) >= 64u) continue;   // block-uniform
            const __bf16* asrc0 = arow0 + j * CIN + cbase;
            const __bf16* asrc1 = arow1 + j * CIN + cbase;
            const __bf16* srow  = &sb[dyi * 64 * PSTR];
            int  xx[4];
            bool bv[4];
            #pragma unroll
            for (int t = 0; t < 4; ++t) {
                xx[t] = t * 16 + lm + dx;
                bv[t] = (unsigned)xx[t] < 64u;
            }
            #pragma unroll
            for (int c0 = 0; c0 < 64; c0 += 32) {
                v8bf a0lo = *(const v8bf*)(asrc0 + c0);
                v8bf a0hi = *(const v8bf*)(asrc0 + c0 + 16);
                v16bf a0  = SHUF16(a0lo, a0hi);
                v8bf a1lo = *(const v8bf*)(asrc1 + c0);
                v8bf a1hi = *(const v8bf*)(asrc1 + c0 + 16);
                v16bf a1  = SHUF16(a1lo, a1hi);
                #pragma unroll
                for (int t = 0; t < 4; ++t) {
                    v16bf b = {};
                    if (bv[t]) {
                        const __bf16* bp = srow + xx[t] * PSTR + c0 + hi * 8;
                        v8bf blo = *(const v8bf*)(bp);
                        v8bf bhi = *(const v8bf*)(bp + 16);
                        b = SHUF16(blo, bhi);
                    }
                    acc[t]     = __builtin_amdgcn_wmma_f32_16x16x32_bf16(
                                     false, a0, false, b, (short)0, acc[t],     false, false);
                    acc[4 + t] = __builtin_amdgcn_wmma_f32_16x16x32_bf16(
                                     false, a1, false, b, (short)0, acc[4 + t], false, false);
                }
            }
        }
    }

    // C/D layout: lane col = lane&15; VGPR r -> row r (+8 for lanes 16..31).
    int mof = hi << 3;
    #pragma unroll
    for (int m = 0; m < 2; ++m) {
        #pragma unroll
        for (int r = 0; r < 8; ++r) {
            int o = o0 + m * 16 + mof + r;
            float sc = scale[o], bi = bias[o];
            float* orow = out + (size_t)(n * COUT + o) * HWP + y * WWD + lm;
            #pragma unroll
            for (int t = 0; t < 4; ++t) {
                float val = acc[m * 4 + t][r] * sc + bi;
                orow[t * 16] = fmaxf(val, 0.0f);
            }
        }
    }
}

// ---------------- attention weights: softmax over 25 shifts ----------------
__global__ void k_attn(const float* __restrict__ e2, float* __restrict__ attw) {
    int tid = blockIdx.x * blockDim.x + threadIdx.x;   // 2*4096
    int n = tid >> 12;
    int pix = tid & 4095;
    int h = pix >> 6, w = pix & 63;
    const float* c2p = e2 + (size_t)((2 + n) * C2) * HWP + pix;
    const float* op  = e2 + (size_t)(n * C2) * HWP;

    float cross[25], y2[25];
    #pragma unroll
    for (int o = 0; o < 25; ++o) { cross[o] = 0.f; y2[o] = 0.f; }
    float x2 = 0.f;

    for (int c = 0; c < C2; ++c) {
        float cv = c2p[(size_t)c * HWP];
        x2 += cv * cv;
        const float* ob = op + (size_t)c * HWP;
        #pragma unroll
        for (int o = 0; o < 25; ++o) {
            int dy = o / 5 - 2, dx = o % 5 - 2;
            int yy = h + dy, xx = w + dx;
            float ov = ((unsigned)yy < 64u && (unsigned)xx < 64u) ? ob[yy * 64 + xx] : 0.f;
            cross[o] += cv * ov;
            y2[o]    += ov * ov;
        }
    }

    float s[25], mx = -3.4e38f;
    #pragma unroll
    for (int o = 0; o < 25; ++o) {
        int dy = o / 5 - 2, dx = o % 5 - 2;
        int yy = h + dy, xx = w + dx;
        bool inb = ((unsigned)yy < 64u) && ((unsigned)xx < 64u);
        float dist = x2 + (inb ? y2[o] : 1e20f) - 2.f * (inb ? cross[o] : 0.f);
        s[o] = 1.f / (dist + 1e-5f);
        mx = fmaxf(mx, s[o]);
    }
    float sum = 0.f;
    #pragma unroll
    for (int o = 0; o < 25; ++o) { s[o] = expf(s[o] - mx); sum += s[o]; }
    float inv = 1.f / sum;
    #pragma unroll
    for (int o = 0; o < 25; ++o) attw[(size_t)o * 8192 + tid] = s[o] * inv;
}

// ---------------- warp + mean with center image ----------------------------
__global__ void k_warp(const float* __restrict__ es, const float* __restrict__ attw,
                       float* __restrict__ fea) {
    int tid = blockIdx.x * blockDim.x + threadIdx.x;   // 2*256*4096
    int pix = tid & 4095;
    int c = (tid >> 12) & 255;
    int n = tid >> 20;
    int h = pix >> 6, w = pix & 63;
    const float* ob = es + (size_t)(n * CS + c) * HWP;
    const float* wb = attw + (size_t)n * HWP + pix;
    float acc = 0.f;
    #pragma unroll
    for (int o = 0; o < 25; ++o) {
        int dy = o / 5 - 2, dx = o % 5 - 2;
        int yy = h + dy, xx = w + dx;
        float ov = ((unsigned)yy < 64u && (unsigned)xx < 64u) ? ob[yy * 64 + xx] : 0.f;
        acc += ov * wb[(size_t)o * 8192];
    }
    float cimg = es[(size_t)((2 + n) * CS + c) * HWP + pix];
    fea[tid] = 0.5f * (cimg + acc * (1.0f / 25.0f));
}

// ---------------- 1x1 classifier ------------------------------------------
__global__ void k_logits(const float* __restrict__ fea, const float* __restrict__ lw,
                         const float* __restrict__ lb, float* __restrict__ xout) {
    int tid = blockIdx.x * blockDim.x + threadIdx.x;   // 2*124*4096
    int pix = tid & 4095;
    int t = tid >> 12;
    int oc = t % NCLS;
    int n  = t / NCLS;
    const float* fb = fea + (size_t)n * CS * HWP + pix;
    const float* wr = lw + oc * CS;
    float acc = lb[oc];
    #pragma unroll 8
    for (int c = 0; c < CS; ++c) acc += fb[(size_t)c * HWP] * wr[c];
    xout[tid] = acc;
}

// ---------------- log_softmax over classes (in place) ----------------------
__global__ void k_logsm(float* __restrict__ xout) {
    int tid = blockIdx.x * blockDim.x + threadIdx.x;   // 2*4096
    int n = tid >> 12, pix = tid & 4095;
    float* p = xout + (size_t)n * NCLS * HWP + pix;
    float mx = -3.4e38f;
    for (int o = 0; o < NCLS; ++o) mx = fmaxf(mx, p[(size_t)o * HWP]);
    float sum = 0.f;
    for (int o = 0; o < NCLS; ++o) sum += expf(p[(size_t)o * HWP] - mx);
    float lse = mx + logf(sum);
    for (int o = 0; o < NCLS; ++o) p[(size_t)o * HWP] -= lse;
}

// ---------------------------------------------------------------------------
extern "C" void kernel_launch(void* const* d_in, const int* in_sizes, int n_in,
                              void* d_out, int out_size, void* d_ws, size_t ws_size,
                              hipStream_t stream) {
    (void)in_sizes; (void)n_in; (void)out_size; (void)ws_size;

    const float* clip   = (const float*)d_in[0];
    const float* embW   = (const float*)d_in[2];
    const float* emb_g  = (const float*)d_in[3];
    const float* emb_b  = (const float*)d_in[4];
    const float* emb_m  = (const float*)d_in[5];
    const float* emb_v  = (const float*)d_in[6];
    const float* emb2W  = (const float*)d_in[7];
    const float* emb2_g = (const float*)d_in[8];
    const float* emb2_b = (const float*)d_in[9];
    const float* emb2_m = (const float*)d_in[10];
    const float* emb2_v = (const float*)d_in[11];
    const float* lastW  = (const float*)d_in[12];
    const float* lastb  = (const float*)d_in[13];

    char* ws = (char*)d_ws;
    __hip_bfloat16* xbf    = (__hip_bfloat16*)(ws);               // 16,777,216 B
    __hip_bfloat16* wbf2   = (__hip_bfloat16*)(ws + 16777216);    //  1,179,648 B
    __hip_bfloat16* wbfS   = (__hip_bfloat16*)(ws + 17956864);    //  2,359,296 B
    float*          scale2 = (float*)(ws + 20316160);             //        512 B
    float*          bias2  = (float*)(ws + 20316672);             //        512 B
    float*          scaleS = (float*)(ws + 20317184);             //      1,024 B
    float*          biasS  = (float*)(ws + 20318208);             //      1,024 B
    float*          convS  = (float*)(ws + 20319232);             // 16,777,216 B
    float*          attw   = (float*)(ws + 37096448);             //    819,200 B
    float*          fea    = (float*)(ws + 37915648);             //  8,388,608 B
    // total workspace: ~46.3 MB

    float* xout  = (float*)d_out;          // log-softmax output (2,124,64,64)
    float* e2out = (float*)d_out + XOUT;   // clip_emb2 output   (4,128,64,64)

    // prep
    k_prep_x<<<32768, 256, 0, stream>>>(clip, xbf);
    k_prep_w<<<2304, 256, 0, stream>>>(emb2W, wbf2, C2);
    k_prep_w<<<4608, 256, 0, stream>>>(embW, wbfS, CS);
    k_prep_bn<<<1, 256, 0, stream>>>(emb2_g, emb2_b, emb2_m, emb2_v, scale2, bias2, C2);
    k_prep_bn<<<1, 256, 0, stream>>>(emb_g, emb_b, emb_m, emb_v, scaleS, biasS, CS);

    // WMMA convs: one workgroup per image row (256 rows), COUT/32 waves each.
    // conv2 writes directly into the clip_emb2 slice of d_out.
    k_conv_wmma<C2><<<256, 128, 0, stream>>>((const __bf16*)xbf, (const __bf16*)wbf2,
                                             scale2, bias2, e2out);
    k_conv_wmma<CS><<<256, 256, 0, stream>>>((const __bf16*)xbf, (const __bf16*)wbfS,
                                             scaleS, biasS, convS);

    // attention / warp / classifier (fp32 VALU, tiny)
    k_attn<<<32, 256, 0, stream>>>(e2out, attw);
    k_warp<<<8192, 256, 0, stream>>>(convS, attw, fea);
    k_logits<<<3968, 256, 0, stream>>>(fea, lastW, lastb, xout);
    k_logsm<<<32, 256, 0, stream>>>(xout);
}
